// RNN_MoW_37349035606295
// MI455X (gfx1250) — matrix operand order repaired and checked by reading, not compiled, not run on hardware
//
#include <hip/hip_runtime.h>
#include <hip/hip_bf16.h>

// ---------------------------------------------------------------------------
// Mixture-of-Weights LSTM for MI455X (gfx1250, wave32, WMMA bf16 + TDM)
//   T=128, B=64, I=H=1024, G=4H=4096, K=8
// ---------------------------------------------------------------------------

#define T_DIM 128
#define B_DIM 64
#define I_DIM 1024
#define H_DIM 1024
#define G_DIM 4096
#define K_DIM 8

// LDS staging for one W_hh tile: 16 rows x 1024 bf16, padded 16B per 1024B
// interval by the TDM (pad_interval=7 -> 256 dw, pad_amount=3 -> 4 dw).
#define ROW_BYTES 2080            // 2048 data + 2*16 pad
#define BUF_BYTES (16 * ROW_BYTES) // 33280, 16B aligned

typedef __attribute__((ext_vector_type(16))) __bf16 bf16x16;
typedef __attribute__((ext_vector_type(8)))  __bf16 bf16x8;
typedef __attribute__((ext_vector_type(8)))  float  f32x8;
typedef __attribute__((ext_vector_type(4)))  unsigned int u32x4;
typedef __attribute__((ext_vector_type(8)))  int i32x8;
typedef __attribute__((ext_vector_type(4)))  int i32x4;

// ---- WMMA fragment loaders (contiguous k = 16*half + s, same for A and B) ------------------
// A: 16x32 bf16, row-major source. lane(0..15)=M, half selects K chunk:
//   half 0: K 0..15 ; half 1: K 16..31
__device__ __forceinline__ bf16x16 load_a_frag(const __bf16* __restrict__ base, int ld) {
  const int lane = threadIdx.x & 31;
  const int m = lane & 15;
  const int half = lane >> 4;
  const __bf16* p = base + (size_t)m * ld + half * 16;
  bf16x8 lo = *(const bf16x8*)(p);          // K = half*16 .. +7
  bf16x8 hi = *(const bf16x8*)(p + 8);      // K = half*16 + 8 .. +15
  return __builtin_shufflevector(lo, hi, 0,1,2,3,4,5,6,7,8,9,10,11,12,13,14,15);
}

// B: 32x16 bf16 where B[kk][n] = W[n][kk]; W row-major so column n of B is a
// contiguous row of W. lane(0..15)=N, half selects K 0..15 vs 16..31.
__device__ __forceinline__ bf16x16 load_b_frag(const __bf16* __restrict__ base, int ld) {
  const int lane = threadIdx.x & 31;
  const int n = lane & 15;
  const int half = lane >> 4;
  const __bf16* p = base + (size_t)n * ld + half * 16;
  bf16x8 lo = *(const bf16x8*)(p);
  bf16x8 hi = *(const bf16x8*)(p + 8);
  return __builtin_shufflevector(lo, hi, 0,1,2,3,4,5,6,7,8,9,10,11,12,13,14,15);
}

// Same B fragment but from the TDM-staged, padded LDS tile.
__device__ __forceinline__ bf16x16 load_b_frag_lds(const unsigned char* bufbase,
                                                   int kkbase) {
  const int lane = threadIdx.x & 31;
  const int n = lane & 15;
  const int half = lane >> 4;
  const int kk = kkbase + half * 16;
  const unsigned char* p =
      bufbase + n * ROW_BYTES + kk * 2 + ((kk >> 9) << 4);  // +16B pad past 512 elems
  bf16x8 lo = *(const bf16x8*)(p);
  bf16x8 hi = *(const bf16x8*)(p + 16);
  return __builtin_shufflevector(lo, hi, 0,1,2,3,4,5,6,7,8,9,10,11,12,13,14,15);
}

__device__ __forceinline__ float sigmoidf_(float x) {
  return 1.0f / (1.0f + __expf(-x));
}

// ---- TDM issue: stage 16 x 1024 bf16 rows of W (rows n0..n0+15 of matrix k)
// into LDS at byte offset lds_off. Descriptor per CDNA5 ISA ch.8.
__device__ __forceinline__ void tdm_load_b_tile(const __bf16* wk_rows,
                                                unsigned int lds_off) {
  const unsigned long long ga = (unsigned long long)(size_t)wk_rows;
  u32x4 g0;
  g0[0] = 1u;                                   // count=1, user descriptor
  g0[1] = lds_off;                              // lds_addr (bytes)
  g0[2] = (unsigned int)(ga & 0xFFFFFFFFu);     // global_addr[31:0]
  g0[3] = (unsigned int)((ga >> 32) & 0x1FFFFFFu) | (2u << 30);  // addr hi | type=2

  i32x8 g1;
  g1[0] = (int)((1u << 16)        // data_size = 1 -> 2 bytes
              | (1u << 20)        // pad_enable
              | (7u << 22)        // pad_interval = 256 dwords (1024B)
              | (3u << 25));      // pad_amount = 4 dwords (16B)
  g1[1] = (int)(1024u << 16);     // tensor_dim0 = 1024 (low 16 into bits 63:48)
  g1[2] = (int)(16u << 16);       // tensor_dim0 hi=0 | tensor_dim1 = 16
  g1[3] = (int)(1024u << 16);     // tensor_dim1 hi=0 | tile_dim0 = 1024
  g1[4] = 16;                     // tile_dim1 = 16, tile_dim2 = 0
  g1[5] = 1024;                   // tensor_dim0_stride low 32
  g1[6] = 0;                      // stride hi | tensor_dim1_stride lo
  g1[7] = 0;

  i32x4 z4; z4[0] = 0; z4[1] = 0; z4[2] = 0; z4[3] = 0;
  i32x8 z8;
#pragma unroll
  for (int i = 0; i < 8; ++i) z8[i] = 0;
  // 6-arg form on this toolchain: (g0, g1, g2, g3, g4, cpol)
  __builtin_amdgcn_tensor_load_to_lds(g0, g1, z4, z4, z8, 0);
}

// ---- tiny prep kernels ------------------------------------------------------
__global__ void softmax_coef_kernel(const float* __restrict__ coef,
                                    float* __restrict__ cw) {
  const int t = threadIdx.x;            // one thread per timestep, T=128
  if (t >= T_DIM) return;
  float v[K_DIM];
  float m = -3.4e38f;
  for (int k = 0; k < K_DIM; ++k) { v[k] = coef[t * K_DIM + k]; m = fmaxf(m, v[k]); }
  float s = 0.0f;
  for (int k = 0; k < K_DIM; ++k) { v[k] = __expf(v[k] - m); s += v[k]; }
  const float inv = 1.0f / s;
  for (int k = 0; k < K_DIM; ++k) cw[t * K_DIM + k] = v[k] * inv;
}

__global__ void bias_mix_kernel(const float* __restrict__ cw,
                                const float* __restrict__ b_ih,
                                const float* __restrict__ b_hh,
                                float* __restrict__ bsum) {
  const int idx = blockIdx.x * blockDim.x + threadIdx.x;  // T*G
  if (idx >= T_DIM * G_DIM) return;
  const int t = idx / G_DIM;
  const int g = idx - t * G_DIM;
  float acc = 0.0f;
  for (int k = 0; k < K_DIM; ++k)
    acc += cw[t * K_DIM + k] * (b_ih[k * G_DIM + g] + b_hh[k * G_DIM + g]);
  bsum[idx] = acc;
}

__global__ void f32_to_bf16_kernel(const float* __restrict__ src,
                                   __bf16* __restrict__ dst, int n) {
  const int idx = blockIdx.x * blockDim.x + threadIdx.x;
  if (idx < n) dst[idx] = (__bf16)src[idx];
}

__global__ void init_state_kernel(const float* __restrict__ h0,
                                  const float* __restrict__ c0,
                                  float* __restrict__ cbuf,
                                  __bf16* __restrict__ hbf) {
  const int idx = blockIdx.x * blockDim.x + threadIdx.x;  // B*H
  if (idx >= B_DIM * H_DIM) return;
  cbuf[idx] = c0[idx];
  hbf[idx] = (__bf16)h0[idx];
}

// ---- phase 1: xproj[t,b,g] = sum_k cw[t,k] * (x_t @ W_ih,k^T) + bias[t,g] ---
// grid: T * (G/32) blocks of 128 threads (4 waves). wave w -> rows 16w..16w+15,
// each wave computes 16x32 (two N-tiles, two independent WMMA chains).
__global__ void xproj_gemm_kernel(const __bf16* __restrict__ xbf,
                                  const __bf16* __restrict__ wihb,
                                  const float* __restrict__ cw,
                                  const float* __restrict__ bsum,
                                  float* __restrict__ xproj) {
  const int t = blockIdx.x >> 7;               // / (G/32)
  const int ntile = blockIdx.x & 127;
  const int n0 = ntile * 32;
  const int w = threadIdx.x >> 5;              // wave id 0..3
  const int lane = threadIdx.x & 31;

  const __bf16* abase = xbf + ((size_t)t * B_DIM + w * 16) * I_DIM;

  f32x8 tot0 = {}, tot1 = {};
  for (int k = 0; k < K_DIM; ++k) {
    const __bf16* bbase = wihb + ((size_t)k * G_DIM + n0) * I_DIM;
    if (k + 1 < K_DIM)
      __builtin_prefetch(wihb + ((size_t)(k + 1) * G_DIM + n0) * I_DIM, 0, 0);
    f32x8 acc0 = {}, acc1 = {};
#pragma unroll 2
    for (int kk = 0; kk < I_DIM; kk += 32) {
      bf16x16 a  = load_a_frag(abase + kk, I_DIM);
      bf16x16 b0 = load_b_frag(bbase + kk, I_DIM);
      bf16x16 b1 = load_b_frag(bbase + (size_t)16 * I_DIM + kk, I_DIM);
      acc0 = __builtin_amdgcn_wmma_f32_16x16x32_bf16(
          false, a, false, b0, (short)0, acc0, false, false);
      acc1 = __builtin_amdgcn_wmma_f32_16x16x32_bf16(
          false, a, false, b1, (short)0, acc1, false, false);
    }
    const float s = cw[t * K_DIM + k];
#pragma unroll
    for (int r = 0; r < 8; ++r) {
      tot0[r] = fmaf(s, acc0[r], tot0[r]);
      tot1[r] = fmaf(s, acc1[r], tot1[r]);
    }
  }

  // C/D layout: lane(0..15)=N, lanes 16..31 -> M+8 ; vgpr r -> M=r
  const int n = lane & 15;
  const int mh = lane >> 4;
#pragma unroll
  for (int r = 0; r < 8; ++r) {
    const int row = w * 16 + mh * 8 + r;
    xproj[((size_t)t * B_DIM + row) * G_DIM + (n0 + n)] =
        tot0[r] + bsum[t * G_DIM + (n0 + n)];
    xproj[((size_t)t * B_DIM + row) * G_DIM + (n0 + 16 + n)] =
        tot1[r] + bsum[t * G_DIM + (n0 + 16 + n)];
  }
}

// ---- phase 2a (per step): gates = xproj[t] + sum_k cw[t,k]*(h @ W_hh,k^T) ---
// grid: G/16 = 256 blocks of 128 threads (4 waves). wave w -> rows 16w..16w+15.
// W_hh tile (16 x 1024 bf16) staged to LDS by the Tensor Data Mover,
// double-buffered; one wave issues TDM, all waves consume via ds_load.
__global__ void step_gemm_kernel(const __bf16* __restrict__ hbf,
                                 const __bf16* __restrict__ whhb,
                                 const float* __restrict__ cw,
                                 const float* __restrict__ xproj,
                                 float* __restrict__ gates, int t) {
  __shared__ __attribute__((aligned(16))) unsigned char smem[2 * BUF_BYTES];

  const int n0 = blockIdx.x * 16;
  const int w = threadIdx.x >> 5;
  const int lane = threadIdx.x & 31;
  // wave-uniform (scalar) guard so only wave 0's stream contains TDM ops
  const bool tdm_wave = (__builtin_amdgcn_readfirstlane(threadIdx.x >> 5) == 0);

  const __bf16* abase = hbf + (size_t)(w * 16) * H_DIM;
  const unsigned int lds_base = (unsigned int)(size_t)(&smem[0]);

  if (tdm_wave) {
    tdm_load_b_tile(whhb + ((size_t)0 * G_DIM + n0) * H_DIM, lds_base);
  }

  f32x8 tot = {};
#pragma unroll
  for (int k = 0; k < K_DIM; ++k) {
    if (tdm_wave) {
      if (k + 1 < K_DIM) {
        tdm_load_b_tile(whhb + ((size_t)(k + 1) * G_DIM + n0) * H_DIM,
                        lds_base + (unsigned int)(((k + 1) & 1) * BUF_BYTES));
        __builtin_amdgcn_s_wait_tensorcnt(1);  // buffer k complete (in-order)
      } else {
        __builtin_amdgcn_s_wait_tensorcnt(0);
      }
    }
    __syncthreads();  // buffer k visible to all waves

    const unsigned char* bbuf = smem + (k & 1) * BUF_BYTES;
    f32x8 acc = {};
#pragma unroll 4
    for (int kk = 0; kk < H_DIM; kk += 32) {
      bf16x16 a = load_a_frag(abase + kk, H_DIM);
      bf16x16 b = load_b_frag_lds(bbuf, kk);
      acc = __builtin_amdgcn_wmma_f32_16x16x32_bf16(
          false, a, false, b, (short)0, acc, false, false);
    }
    const float s = cw[t * K_DIM + k];
#pragma unroll
    for (int r = 0; r < 8; ++r) tot[r] = fmaf(s, acc[r], tot[r]);

    __syncthreads();  // all reads done before buffer (k&1) is overwritten at k+2
  }

  const int n = lane & 15;
  const int mh = lane >> 4;
#pragma unroll
  for (int r = 0; r < 8; ++r) {
    const int row = w * 16 + mh * 8 + r;
    const int col = n0 + n;
    gates[(size_t)row * G_DIM + col] =
        tot[r] + xproj[((size_t)t * B_DIM + row) * G_DIM + col];
  }
}

// ---- phase 2b (per step): LSTM pointwise update -----------------------------
__global__ void step_pointwise_kernel(const float* __restrict__ gates,
                                      float* __restrict__ cbuf,
                                      __bf16* __restrict__ hbf,
                                      float* __restrict__ out, int t) {
  const int idx = blockIdx.x * blockDim.x + threadIdx.x;  // B*H
  if (idx >= B_DIM * H_DIM) return;
  const int b = idx >> 10;
  const int g = idx & (H_DIM - 1);
  const float* gr = gates + (size_t)b * G_DIM;
  const float iv = sigmoidf_(gr[g]);
  const float fv = sigmoidf_(gr[H_DIM + g]);
  const float gv = tanhf(gr[2 * H_DIM + g]);
  const float ov = sigmoidf_(gr[3 * H_DIM + g]);
  const float cn = fv * cbuf[idx] + iv * gv;
  cbuf[idx] = cn;
  const float hn = ov * tanhf(cn);
  out[(size_t)t * (B_DIM * H_DIM) + idx] = hn;
  hbf[idx] = (__bf16)hn;
}

// ---------------------------------------------------------------------------
static inline size_t align256(size_t x) { return (x + 255) & ~(size_t)255; }

extern "C" void kernel_launch(void* const* d_in, const int* in_sizes, int n_in,
                              void* d_out, int out_size, void* d_ws, size_t ws_size,
                              hipStream_t stream) {
  const float* x    = (const float*)d_in[0];   // [T,B,I]
  const float* coef = (const float*)d_in[1];   // [T,K]
  const float* w_ih = (const float*)d_in[2];   // [K,G,I]
  const float* w_hh = (const float*)d_in[3];   // [K,G,H]
  const float* b_ih = (const float*)d_in[4];   // [K,G]
  const float* b_hh = (const float*)d_in[5];   // [K,G]
  const float* h0   = (const float*)d_in[6];   // [B,H]
  const float* c0   = (const float*)d_in[7];   // [B,H]
  float* out = (float*)d_out;                  // [T,B,H]

  // workspace carve-up
  char* ws = (char*)d_ws;
  size_t off = 0;
  float*  cw    = (float*)(ws + off); off = align256(off + (size_t)T_DIM * K_DIM * 4);
  float*  bsum  = (float*)(ws + off); off = align256(off + (size_t)T_DIM * G_DIM * 4);
  __bf16* wihb  = (__bf16*)(ws + off); off = align256(off + (size_t)K_DIM * G_DIM * I_DIM * 2);
  __bf16* whhb  = (__bf16*)(ws + off); off = align256(off + (size_t)K_DIM * G_DIM * H_DIM * 2);
  __bf16* xbf   = (__bf16*)(ws + off); off = align256(off + (size_t)T_DIM * B_DIM * I_DIM * 2);
  float*  xproj = (float*)(ws + off); off = align256(off + (size_t)T_DIM * B_DIM * G_DIM * 4);
  float*  gates = (float*)(ws + off); off = align256(off + (size_t)B_DIM * G_DIM * 4);
  float*  cbuf  = (float*)(ws + off); off = align256(off + (size_t)B_DIM * H_DIM * 4);
  __bf16* hbf   = (__bf16*)(ws + off); off = align256(off + (size_t)B_DIM * H_DIM * 2);
  (void)ws_size; (void)in_sizes; (void)n_in; (void)out_size;

  // prep
  softmax_coef_kernel<<<1, 128, 0, stream>>>(coef, cw);
  bias_mix_kernel<<<(T_DIM * G_DIM) / 256, 256, 0, stream>>>(cw, b_ih, b_hh, bsum);

  const int n_wih = K_DIM * G_DIM * I_DIM;
  const int n_whh = K_DIM * G_DIM * H_DIM;
  const int n_x   = T_DIM * B_DIM * I_DIM;
  f32_to_bf16_kernel<<<(n_wih + 255) / 256, 256, 0, stream>>>(w_ih, wihb, n_wih);
  f32_to_bf16_kernel<<<(n_whh + 255) / 256, 256, 0, stream>>>(w_hh, whhb, n_whh);
  f32_to_bf16_kernel<<<(n_x + 255) / 256, 256, 0, stream>>>(x, xbf, n_x);
  init_state_kernel<<<(B_DIM * H_DIM) / 256, 256, 0, stream>>>(h0, c0, cbuf, hbf);

  // batched input projections with k-mixture folded into the epilogue
  xproj_gemm_kernel<<<T_DIM * (G_DIM / 32), 128, 0, stream>>>(xbf, wihb, cw, bsum, xproj);

  // sequential recurrence
  for (int t = 0; t < T_DIM; ++t) {
    step_gemm_kernel<<<G_DIM / 16, 128, 0, stream>>>(hbf, whhb, cw, xproj, gates, t);
    step_pointwise_kernel<<<(B_DIM * H_DIM) / 256, 256, 0, stream>>>(gates, cbuf, hbf, out, t);
  }
}
